// RelationalMemoryNeuro_21698174779511
// MI455X (gfx1250) — matrix operand order repaired and checked by reading, not compiled, not run on hardware
//
#include <hip/hip_runtime.h>
#include <hip/hip_bf16.h>
#include <stdint.h>

// ---------------------------------------------------------------------------
// RelationalMemoryNeuro for MI455X (gfx1250, wave32, WMMA + TDM)
//   scores GEMM: bf16x3 split through V_WMMA_F32_16X16X32_BF16, tiles staged
//                by the Tensor Data Mover (tensor_load_to_lds, TENSORcnt,
//                hardware LDS padding for bank-conflict-free fragments),
//                double-buffered for DMA/compute overlap.
//   top-k:       per-row LDS radix-select + softmax (+active-concept renorm)
//   context:     per-token low-rank relation math + proto gather + residual
// ---------------------------------------------------------------------------

typedef __bf16 bf16_t;
typedef __attribute__((ext_vector_type(16))) __bf16 v16bf;
typedef __attribute__((ext_vector_type(8)))  float  v8f;
typedef __attribute__((ext_vector_type(4)))  unsigned int v4u;
typedef __attribute__((ext_vector_type(4)))  int v4i;
typedef __attribute__((ext_vector_type(8)))  int v8i;

#define D_DIM   1024
#define N_CON   4096
#define R_RANK  16
#define N_REL   8
#define TOPK    128
#define SCALE_INV (1.0f / 32.0f)   // 1/sqrt(1024)

// ---------------------------------------------------------------- split fp32 -> bf16 hi + bf16 lo
__global__ void split_bf16_kernel(const float* __restrict__ src,
                                  bf16_t* __restrict__ hi,
                                  bf16_t* __restrict__ lo, int n) {
  int i = blockIdx.x * blockDim.x + threadIdx.x;
  if (i < n) {
    float x = src[i];
    bf16_t h = (bf16_t)x;
    hi[i] = h;
    lo[i] = (bf16_t)(x - (float)h);
  }
}

// ---------------------------------------------------------------- any(concept_used)
__global__ void any_used_kernel(const int* __restrict__ used, int n, int* __restrict__ flag) {
  __shared__ int s;
  if (threadIdx.x == 0) s = 0;
  __syncthreads();
  int acc = 0;
  for (int i = threadIdx.x; i < n; i += blockDim.x) acc |= (used[i] != 0);
  if (acc) atomicOr(&s, 1);
  __syncthreads();
  if (threadIdx.x == 0) *flag = s;
}

// ---------------------------------------------------------------- TDM 2D tile load: global -> LDS
// D# per CDNA5 ISA. 2D tile, bf16 (data_size=1),
// LDS padding: 1 DWORD after every 16 DWORDs (row stride 32 bf16 -> 34 bf16).
__device__ __forceinline__ void tdm_load_tile_bf16(unsigned lds_off, const bf16_t* gptr,
                                                   unsigned tile0, unsigned tile1,
                                                   unsigned long long elem_stride0) {
  unsigned long long ga = (unsigned long long)(uintptr_t)gptr;
  v4u g0;
  g0[0] = 1u;                                            // count=1, user descriptor
  g0[1] = lds_off;                                       // lds_addr [63:32]
  g0[2] = (unsigned)(ga & 0xFFFFFFFFu);                  // global_addr lo
  g0[3] = (unsigned)((ga >> 32) & 0x01FFFFFFu)           // global_addr hi (57-bit)
          | 0x80000000u;                                 // type=2 ("image") [127:126]
  v8i g1;
  // [15:0] wg_mask=0 | [17:16] data_size=1 (2B) | [20] pad_enable=1
  // [24:22] pad_interval=3 (16 DWORDs) | [31:25] pad_amount=0 (1 DWORD)
  g1[0] = (int)((1u << 16) | (1u << 20) | (3u << 22));
  g1[1] = (int)((tile0 & 0xFFFFu) << 16);                // tensor_dim0[15:0] (== tile0)
  g1[2] = (int)(((tile0 >> 16) & 0xFFFFu) | ((tile1 & 0xFFFFu) << 16)); // t_dim0 hi | t_dim1 lo
  g1[3] = (int)(((tile1 >> 16) & 0xFFFFu) | ((tile0 & 0xFFFFu) << 16)); // t_dim1 hi | tile_dim0
  g1[4] = (int)(tile1 & 0xFFFFu);                        // tile_dim1 | tile_dim2=0
  g1[5] = (int)(elem_stride0 & 0xFFFFFFFFull);           // tensor_dim0_stride lo
  g1[6] = (int)((elem_stride0 >> 32) & 0xFFFFull);       // stride hi | dim1_stride lo = 0
  g1[7] = 0;
  v4i z4 = {0, 0, 0, 0};
#if __clang_major__ >= 23
  v8i z8 = {0, 0, 0, 0, 0, 0, 0, 0};
  __builtin_amdgcn_tensor_load_to_lds(g0, g1, z4, z4, z8, 0);
#else
  __builtin_amdgcn_tensor_load_to_lds(g0, g1, z4, z4, 0);
#endif
}

// ---------------------------------------------------------------- scores = (x @ proto^T)/32  via WMMA bf16x3
__global__ __launch_bounds__(256)
void scores_wmma_kernel(const bf16_t* __restrict__ xhi, const bf16_t* __restrict__ xlo,
                        const bf16_t* __restrict__ phi, const bf16_t* __restrict__ plo,
                        float* __restrict__ scores) {
  constexpr int BM = 64, BN = 128, BK = 32, LSTR = 34;  // 34 bf16 = 17 DWORD row stride (TDM pad)
  __shared__ bf16_t sa_hi[2][BM][LSTR];
  __shared__ bf16_t sa_lo[2][BM][LSTR];
  __shared__ bf16_t sb_hi[2][BN][LSTR];
  __shared__ bf16_t sb_lo[2][BN][LSTR];

  const int tid  = threadIdx.x;
  const int lane = tid & 31;
  const int wave = tid >> 5;
  const int wm   = wave & 1;    // 2 M-subtiles of 32
  const int wn   = wave >> 1;   // 4 N-subtiles of 32
  const int m0   = blockIdx.y * BM;
  const int n0   = blockIdx.x * BN;

  const int lhalf = (lane >= 16) ? 1 : 0;
  const int lmod  = lane & 15;

  v8f acc[2][2] = {};

  // waves 0..3 each own one TDM stream (x_hi / x_lo / p_hi / p_lo)
  auto issue = [&](int buf, int kb) {
    if (wave == 0)
      tdm_load_tile_bf16((unsigned)(uintptr_t)&sa_hi[buf][0][0],
                         xhi + (size_t)m0 * D_DIM + kb, BK, BM, D_DIM);
    else if (wave == 1)
      tdm_load_tile_bf16((unsigned)(uintptr_t)&sa_lo[buf][0][0],
                         xlo + (size_t)m0 * D_DIM + kb, BK, BM, D_DIM);
    else if (wave == 2)
      tdm_load_tile_bf16((unsigned)(uintptr_t)&sb_hi[buf][0][0],
                         phi + (size_t)n0 * D_DIM + kb, BK, BN, D_DIM);
    else if (wave == 3)
      tdm_load_tile_bf16((unsigned)(uintptr_t)&sb_lo[buf][0][0],
                         plo + (size_t)n0 * D_DIM + kb, BK, BN, D_DIM);
  };

  auto compute = [&](int buf) {
    // fragments use contiguous k order, identical for A and B:
    //  A 16x32: k = e + (lane>=16 ? 16 : 0),                 m = lane%16
    //  B 32x16: k = e + (lane>=16 ? 16 : 0),                 n = lane%16
    v16bf ahi[2], alo[2], bhi[2], blo[2];
#pragma unroll
    for (int i = 0; i < 2; ++i) {
      int mrow = wm * 32 + i * 16 + lmod;
#pragma unroll
      for (int e = 0; e < 16; ++e) {
        int k = e + lhalf * 16;
        ahi[i][e] = sa_hi[buf][mrow][k];
        alo[i][e] = sa_lo[buf][mrow][k];
      }
    }
#pragma unroll
    for (int j = 0; j < 2; ++j) {
      int ncol = wn * 32 + j * 16 + lmod;
#pragma unroll
      for (int e = 0; e < 16; ++e) {
        int k = e + lhalf * 16;
        bhi[j][e] = sb_hi[buf][ncol][k];
        blo[j][e] = sb_lo[buf][ncol][k];
      }
    }
#pragma unroll
    for (int i = 0; i < 2; ++i)
#pragma unroll
      for (int j = 0; j < 2; ++j) {
        acc[i][j] = __builtin_amdgcn_wmma_f32_16x16x32_bf16(false, ahi[i], false, bhi[j],
                                                            (short)0, acc[i][j], false, false);
        acc[i][j] = __builtin_amdgcn_wmma_f32_16x16x32_bf16(false, ahi[i], false, blo[j],
                                                            (short)0, acc[i][j], false, false);
        acc[i][j] = __builtin_amdgcn_wmma_f32_16x16x32_bf16(false, alo[i], false, bhi[j],
                                                            (short)0, acc[i][j], false, false);
      }
  };

  constexpr int nIter = D_DIM / BK;   // 32 K-panels
  issue(0, 0);
  for (int it = 0; it < nIter - 1; ++it) {
    issue((it + 1) & 1, (it + 1) * BK);          // prefetch next panel via TDM
    __builtin_amdgcn_s_wait_tensorcnt(1);        // oldest (current-panel) DMA done
    __syncthreads();                             // LDS visible to all 8 waves
    compute(it & 1);
    __syncthreads();                             // all readers done before overwrite
  }
  __builtin_amdgcn_s_wait_tensorcnt(0);
  __syncthreads();
  compute((nIter - 1) & 1);

  // C/D layout: vgpr v holds M = v + (lane>=16 ? 8 : 0), N = lane%16
#pragma unroll
  for (int i = 0; i < 2; ++i)
#pragma unroll
    for (int j = 0; j < 2; ++j)
#pragma unroll
      for (int v = 0; v < 8; ++v) {
        int m = m0 + wm * 32 + i * 16 + v + lhalf * 8;
        int n = n0 + wn * 32 + j * 16 + lmod;
        scores[(size_t)m * N_CON + n] = acc[i][j][v] * SCALE_INV;
      }
}

// ---------------------------------------------------------------- per-row top-128 + softmax (+mask renorm)
__global__ __launch_bounds__(256)
void topk_softmax_kernel(const float* __restrict__ scores, const int* __restrict__ used,
                         const int* __restrict__ flag_p,
                         int* __restrict__ idx_out, float* __restrict__ att_out) {
  const int row = blockIdx.x;
  const int tid = threadIdx.x;
  const float* srow = scores + (size_t)row * N_CON;

  __shared__ unsigned int hist[16];
  __shared__ unsigned int sh_prefix, sh_remaining;
  __shared__ unsigned int cnt_gt, cnt_eq;
  __shared__ int   sel_idx[TOPK];
  __shared__ float sel_val[TOPK];
  __shared__ float red[256];
  __shared__ int   s_flag;

  // order-preserving float -> uint key
  unsigned int key[16];
#pragma unroll
  for (int j = 0; j < 16; ++j) {
    unsigned int b = __float_as_uint(srow[tid + 256 * j]);
    key[j] = (b & 0x80000000u) ? ~b : (b | 0x80000000u);
  }

  if (tid == 0) { sh_prefix = 0u; sh_remaining = TOPK; cnt_gt = 0u; cnt_eq = 0u; s_flag = *flag_p; }
  __syncthreads();

  // nibble radix-select for the 128th largest key
  for (int p = 7; p >= 0; --p) {
    const int shift = p * 4;
    if (tid < 16) hist[tid] = 0u;
    __syncthreads();
    const unsigned int prefix = sh_prefix;
    const unsigned int himask = (p == 7) ? 0u : (0xFFFFFFFFu << (shift + 4));
#pragma unroll
    for (int j = 0; j < 16; ++j)
      if ((key[j] & himask) == prefix)
        atomicAdd(&hist[(key[j] >> shift) & 0xFu], 1u);
    __syncthreads();
    if (tid == 0) {
      unsigned int rem = sh_remaining, cum = 0u;
      int d = 0;
      for (int dd = 15; dd >= 0; --dd) {
        if (cum + hist[dd] >= rem) { d = dd; break; }
        cum += hist[dd];
      }
      sh_remaining = rem - cum;
      sh_prefix = prefix | ((unsigned int)d << shift);
    }
    __syncthreads();
  }

  const unsigned int thresh = sh_prefix;
  const unsigned int n_ties = sh_remaining;   // # of ==thresh entries to keep

  // compact: strictly-greater first, then ties up to n_ties
#pragma unroll
  for (int j = 0; j < 16; ++j)
    if (key[j] > thresh) {
      unsigned int slot = atomicAdd(&cnt_gt, 1u);
      unsigned int u = key[j];
      unsigned int b = (u & 0x80000000u) ? (u ^ 0x80000000u) : ~u;
      sel_idx[slot] = tid + 256 * j;
      sel_val[slot] = __uint_as_float(b);
    }
  __syncthreads();
  const unsigned int ngt = cnt_gt;
#pragma unroll
  for (int j = 0; j < 16; ++j)
    if (key[j] == thresh) {
      unsigned int t = atomicAdd(&cnt_eq, 1u);
      if (t < n_ties) {
        unsigned int u = key[j];
        unsigned int b = (u & 0x80000000u) ? (u ^ 0x80000000u) : ~u;
        sel_idx[ngt + t] = tid + 256 * j;
        sel_val[ngt + t] = __uint_as_float(b);
      }
    }
  __syncthreads();

  // softmax over the 128 selected values
  red[tid] = (tid < TOPK) ? sel_val[tid] : -3.402823466e38f;
  __syncthreads();
  for (int s = 128; s > 0; s >>= 1) { if (tid < s) red[tid] = fmaxf(red[tid], red[tid + s]); __syncthreads(); }
  const float mx = red[0];
  __syncthreads();
  float e = (tid < TOPK) ? expf(sel_val[tid] - mx) : 0.0f;
  red[tid] = e;
  __syncthreads();
  for (int s = 128; s > 0; s >>= 1) { if (tid < s) red[tid] += red[tid + s]; __syncthreads(); }
  const float sum = red[0];
  __syncthreads();
  float att = e / sum;

  // active-concept masking + renorm (uniform branch)
  if (s_flag) {
    float act = 0.0f;
    if (tid < TOPK) act = (used[sel_idx[tid]] != 0) ? 1.0f : 0.0f;
    float am = att * act;
    red[tid] = am;
    __syncthreads();
    for (int s = 128; s > 0; s >>= 1) { if (tid < s) red[tid] += red[tid + s]; __syncthreads(); }
    const float s2 = red[0];
    __syncthreads();
    att = am / fmaxf(s2, 1e-8f);
  }

  if (tid < TOPK) {
    idx_out[(size_t)row * TOPK + tid] = sel_idx[tid];
    att_out[(size_t)row * TOPK + tid] = att;
  }
}

// ---------------------------------------------------------------- per-token low-rank context + residual
__global__ __launch_bounds__(256)
void context_kernel(const float* __restrict__ x, const float* __restrict__ proto,
                    const float* __restrict__ A, const float* __restrict__ Bm,
                    const float* __restrict__ gains,
                    const int* __restrict__ idx_in, const float* __restrict__ att_in,
                    float* __restrict__ y) {
  const int row = blockIdx.x;
  const int tid = threadIdx.x;
  __shared__ int   s_idx[TOPK];
  __shared__ float s_att[TOPK];
  __shared__ float s_z[N_REL * R_RANK];
  __shared__ float s_c[TOPK];
  __shared__ float s_g[N_REL];
  __shared__ float s_denom;

  if (tid < TOPK) {
    s_idx[tid] = idx_in[(size_t)row * TOPK + tid];
    s_att[tid] = att_in[(size_t)row * TOPK + tid];
  }
  if (tid < N_REL) s_g[tid] = gains[tid];
  if (tid == 0) {
    float ds = 0.0f;
    for (int r = 0; r < N_REL; ++r) ds += gains[r];
    s_denom = (ds <= 0.0f) ? 1.0f : ds;
  }
  __syncthreads();

  // z[r][c] = sum_k att[k] * A[r, idx[k], c]
  if (tid < N_REL * R_RANK) {
    const int r = tid >> 4, c = tid & 15;
    const float* Ar = A + (size_t)r * N_CON * R_RANK;
    float acc = 0.0f;
    for (int k = 0; k < TOPK; ++k) acc += s_att[k] * Ar[(size_t)s_idx[k] * R_RANK + c];
    s_z[tid] = acc;
  }
  __syncthreads();

  // c_total[k] = sum_r gains[r] * (z[r] . B[r, idx[k], :])
  if (tid < TOPK) {
    const int n = s_idx[tid];
    float acc = 0.0f;
#pragma unroll
    for (int r = 0; r < N_REL; ++r) {
      const float* Br = Bm + ((size_t)r * N_CON + n) * R_RANK;
      float part = 0.0f;
#pragma unroll
      for (int c = 0; c < R_RANK; ++c) part += s_z[r * R_RANK + c] * Br[c];
      acc += s_g[r] * part;
    }
    s_c[tid] = acc;
  }
  __syncthreads();

  // ctx[d] = sum_k c_total[k] * proto[idx[k], d];  y = x + 0.1*ctx/denom
  float acc[4] = {0.f, 0.f, 0.f, 0.f};
  for (int k = 0; k < TOPK; ++k) {
    const float c = s_c[k];
    const float* pr = proto + (size_t)s_idx[k] * D_DIM;
#pragma unroll
    for (int j = 0; j < 4; ++j) acc[j] += c * pr[tid + 256 * j];
  }
  const float scl = 0.1f / s_denom;
  const size_t base = (size_t)row * D_DIM;
#pragma unroll
  for (int j = 0; j < 4; ++j) {
    const int d = tid + 256 * j;
    y[base + d] = x[base + d] + acc[j] * scl;
  }
}

// ---------------------------------------------------------------- host launcher
extern "C" void kernel_launch(void* const* d_in, const int* in_sizes, int n_in,
                              void* d_out, int out_size, void* d_ws, size_t ws_size,
                              hipStream_t stream) {
  const float* x     = (const float*)d_in[0];
  const int*   used  = (const int*)d_in[1];
  const float* proto = (const float*)d_in[2];
  const float* A     = (const float*)d_in[3];
  const float* Bm    = (const float*)d_in[4];
  const float* gains = (const float*)d_in[5];
  float* y = (float*)d_out;

  const int BT = in_sizes[0] / D_DIM;   // 2048 tokens (B*T), multiple of 64

  // workspace carve-out
  char* ws = (char*)d_ws;
  size_t off = 0;
  auto carve = [&](size_t bytes) -> char* {
    char* p = ws + off;
    off = (off + bytes + 255) & ~(size_t)255;
    return p;
  };
  bf16_t* xhi    = (bf16_t*)carve((size_t)BT * D_DIM * sizeof(bf16_t));
  bf16_t* xlo    = (bf16_t*)carve((size_t)BT * D_DIM * sizeof(bf16_t));
  bf16_t* phi    = (bf16_t*)carve((size_t)N_CON * D_DIM * sizeof(bf16_t));
  bf16_t* plo    = (bf16_t*)carve((size_t)N_CON * D_DIM * sizeof(bf16_t));
  float*  scores = (float*)carve((size_t)BT * N_CON * sizeof(float));
  int*    idxb   = (int*)carve((size_t)BT * TOPK * sizeof(int));
  float*  attb   = (float*)carve((size_t)BT * TOPK * sizeof(float));
  int*    flag   = (int*)carve(sizeof(int));

  // 1) bf16 hi/lo splits of x and proto
  {
    int nx = BT * D_DIM;
    split_bf16_kernel<<<(nx + 255) / 256, 256, 0, stream>>>(x, xhi, xlo, nx);
    int np = N_CON * D_DIM;
    split_bf16_kernel<<<(np + 255) / 256, 256, 0, stream>>>(proto, phi, plo, np);
  }

  // 2) any(concept_used)
  any_used_kernel<<<1, 256, 0, stream>>>(used, N_CON, flag);

  // 3) scores GEMM via WMMA + TDM double-buffered K panels
  {
    dim3 grid(N_CON / 128, BT / 64);
    scores_wmma_kernel<<<grid, 256, 0, stream>>>(xhi, xlo, phi, plo, scores);
  }

  // 4) top-128 + softmax per row
  topk_softmax_kernel<<<BT, 256, 0, stream>>>(scores, used, flag, idxb, attb);

  // 5) low-rank relation context + residual
  context_kernel<<<BT, 256, 0, stream>>>(x, proto, A, Bm, gains, idxb, attb, y);
}
